// AWQW4A16Linear_35150012350702
// MI455X (gfx1250) — matrix operand order repaired and checked by reading, not compiled, not run on hardware
//
#include <hip/hip_runtime.h>

typedef _Float16 v4h  __attribute__((ext_vector_type(4)));
typedef _Float16 v8h  __attribute__((ext_vector_type(8)));
typedef _Float16 v16h __attribute__((ext_vector_type(16)));
typedef float    v8f  __attribute__((ext_vector_type(8)));

#define M_DIM 4096
#define K_DIM 8192
#define O_DIM 8192
#define BM 128
#define BN 128
#define BK 64
#define LDK 72   // padded row stride in halves (144 B) to stagger LDS banks

union Frag16 { v16h v; v8h h[2]; };

__global__ __launch_bounds__(256)
void awq_w4a16_wmma(const float* __restrict__ x,
                    const int*   __restrict__ qweight,   // one byte value per int32
                    const float* __restrict__ wscales,   // [K/128, O]
                    const float* __restrict__ zeros,     // [K/128, O]
                    float* __restrict__ out)             // [M, O]
{
    __shared__ _Float16 sx[BM * LDK];   // x tile, f16, [m][k]
    __shared__ _Float16 sw[BN * LDK];   // dequantized weight tile, f16, [n][k]

    const int t        = threadIdx.x;
    const int lane     = t & 31;
    const int wave     = t >> 5;
    const int lane16   = lane & 15;
    const int half_sel = lane >> 4;

    const int block_n = blockIdx.x * BN;
    const int block_m = blockIdx.y * BM;

    // 8 waves -> 2 along M (64 rows each) x 4 along N (32 cols each)
    const int wm = (wave >> 2) * 64;
    const int wn = (wave & 3) * 32;

    v8f acc[4][2];
    #pragma unroll
    for (int i = 0; i < 4; ++i)
        #pragma unroll
        for (int j = 0; j < 2; ++j)
            acc[i][j] = (v8f){};

    // per-thread staging coordinates
    const int xrow = t >> 4;           // +16*i  (8 float4 per thread)
    const int xcol = (t & 15) << 2;    // float col 0..60
    const int wrow = t >> 3;           // +32*i  (4 int4 per thread)
    const int wcol = (t & 7) << 3;     // k col 0..56 (8 k-values per int4)

    for (int k0 = 0; k0 < K_DIM; k0 += BK) {
        const int g = k0 >> 7;         // quant group (GROUP_SIZE = 128)

        // ---- prefetch next slab (global_prefetch_b8) ----
        if (k0 + BK < K_DIM) {
            __builtin_prefetch(x + (size_t)(block_m + xrow) * K_DIM + (k0 + BK) + xcol, 0, 1);
            __builtin_prefetch(qweight + (size_t)(block_n + wrow) * (K_DIM / 2)
                                        + (((k0 + BK) + wcol) >> 1), 0, 1);
        }

        // ---- global loads; convert/dequant in registers BEFORE the barrier ----
        v4h xh[8];
        #pragma unroll
        for (int i = 0; i < 8; ++i) {
            const int r = xrow + (i << 4);
            const float4 v = *(const float4*)(x + (size_t)(block_m + r) * K_DIM + k0 + xcol);
            xh[i][0] = (_Float16)v.x; xh[i][1] = (_Float16)v.y;
            xh[i][2] = (_Float16)v.z; xh[i][3] = (_Float16)v.w;
        }
        v8h wh[4];
        #pragma unroll
        for (int i = 0; i < 4; ++i) {
            const int r  = wrow + (i << 5);
            const int og = block_n + r;
            const int4 q4 = *(const int4*)(qweight + (size_t)og * (K_DIM / 2) + ((k0 + wcol) >> 1));
            const float s = wscales[(size_t)g * O_DIM + og];
            const float z = zeros  [(size_t)g * O_DIM + og];
            const int q[4] = { q4.x, q4.y, q4.z, q4.w };
            #pragma unroll
            for (int j = 0; j < 4; ++j) {
                wh[i][2 * j + 0] = (_Float16)((float)( q[j]       & 0xF) * s - z);
                wh[i][2 * j + 1] = (_Float16)((float)((q[j] >> 4) & 0xF) * s - z);
            }
        }

        __syncthreads();   // previous slab fully consumed

        #pragma unroll
        for (int i = 0; i < 8; ++i)
            *(v4h*)(&sx[(xrow + (i << 4)) * LDK + xcol]) = xh[i];
        #pragma unroll
        for (int i = 0; i < 4; ++i)
            *(v8h*)(&sw[(wrow + (i << 5)) * LDK + wcol]) = wh[i];

        __syncthreads();   // slab staged

        // ---- 2 k-steps of 16x16x32 f16 WMMA ----
        #pragma unroll
        for (int ks = 0; ks < 2; ++ks) {
            // A fragment: lanes 0-15 -> K 0-15; lanes 16-31 -> K 16-31 (contiguous)
            const int ka = ks * 32 + half_sel * 16;
            // B fragment: lanes 0-15 -> K 0-15; lanes 16-31 -> K 16-31 (contiguous)
            const int kbb = ks * 32 + half_sel * 16;

            Frag16 a[4], b[2];
            #pragma unroll
            for (int tm = 0; tm < 4; ++tm) {
                const _Float16* p = &sx[(wm + tm * 16 + lane16) * LDK + ka];
                a[tm].h[0] = *(const v8h*)(p);
                a[tm].h[1] = *(const v8h*)(p + 8);
            }
            #pragma unroll
            for (int tn = 0; tn < 2; ++tn) {
                const _Float16* p = &sw[(wn + tn * 16 + lane16) * LDK + kbb];
                b[tn].h[0] = *(const v8h*)(p);
                b[tn].h[1] = *(const v8h*)(p + 8);
            }
            #pragma unroll
            for (int tm = 0; tm < 4; ++tm)
                #pragma unroll
                for (int tn = 0; tn < 2; ++tn)
                    acc[tm][tn] = __builtin_amdgcn_wmma_f32_16x16x32_f16(
                        false, a[tm].v, false, b[tn].v, (short)0, acc[tm][tn], false, false);
        }
    }

    // ---- write back: C/D layout VGPR r -> M = r (+8 for upper lane half) ----
    #pragma unroll
    for (int tm = 0; tm < 4; ++tm) {
        #pragma unroll
        for (int tn = 0; tn < 2; ++tn) {
            const int col  = block_n + wn + tn * 16 + lane16;
            const int row0 = block_m + wm + tm * 16 + half_sel * 8;
            #pragma unroll
            for (int r = 0; r < 8; ++r)
                out[(size_t)(row0 + r) * O_DIM + col] = acc[tm][tn][r];
        }
    }
}

extern "C" void kernel_launch(void* const* d_in, const int* in_sizes, int n_in,
                              void* d_out, int out_size, void* d_ws, size_t ws_size,
                              hipStream_t stream) {
    const float* x  = (const float*)d_in[0];
    const int*   qw = (const int*)  d_in[1];
    const float* ws = (const float*)d_in[2];
    const float* zs = (const float*)d_in[3];
    float* out = (float*)d_out;

    dim3 grid(O_DIM / BN, M_DIM / BM);   // 64 x 32 blocks
    awq_w4a16_wmma<<<grid, 256, 0, stream>>>(x, qw, ws, zs, out);
}
